// AttentionPoolCompressor_81844896792933
// MI455X (gfx1250) — compile-verified, matrix operand order checked by reading, not run on hardware
//
#include <hip/hip_runtime.h>

typedef __attribute__((ext_vector_type(16))) _Float16 v16h;
typedef __attribute__((ext_vector_type(8)))  _Float16 v8h;
typedef __attribute__((ext_vector_type(4)))  _Float16 v4h;
typedef __attribute__((ext_vector_type(8)))  float    v8f;

#define D_MODEL 512
#define RATIO   4
#define M_BLK   32                    // pooled rows per workgroup
#define ROWS_BLK (M_BLK * RATIO)      // 128 chunk rows per workgroup
#define A_LD    520                   // padded f16 leading dim for A tile (1040B rows)
#define KBLK    32
#define NKB     (D_MODEL / KBLK)      // 16 K-steps

// ---------------------------------------------------------------------------
// Pre-pass: convert w [512x512] f32 -> f16 in workspace (512 KB, L2-resident)
// ---------------------------------------------------------------------------
__global__ __launch_bounds__(256) void cvt_w_f16(const float* __restrict__ w,
                                                 _Float16* __restrict__ wf16) {
  int idx = (blockIdx.x * 256 + threadIdx.x) * 4;
  float4 x = *(const float4*)(w + idx);
  v4h h;
  h[0] = (_Float16)x.x; h[1] = (_Float16)x.y;
  h[2] = (_Float16)x.z; h[3] = (_Float16)x.w;
  *(v4h*)(wf16 + idx) = h;
}

// ---------------------------------------------------------------------------
// Fused: single-pass chunk read -> scores + f16 stage (LDS) -> softmax(4)
//        -> pooled A tile (register-aliased LDS) -> 2x2-tiled WMMA GEMM + bias
// ---------------------------------------------------------------------------
__global__ __launch_bounds__(256) void attnpool_wmma(
    const float* __restrict__ chunk, const float* __restrict__ query,
    const _Float16* __restrict__ wf16, const float* __restrict__ bias,
    float* __restrict__ out) {
  // 128 KB staging buffer; the 33 KB pooled A tile aliases its start.
  __shared__ __align__(16) _Float16 sBuf[ROWS_BLK * D_MODEL];
  __shared__ float sScores[ROWS_BLK];
  __shared__ float sAttn[ROWS_BLK];
  _Float16* sC = sBuf;                // f16 chunk tile [128 x 512]
  _Float16* sA = sBuf;                // pooled tile [32 x A_LD], written after barrier

  const int tid  = threadIdx.x;
  const int lane = tid & 31;
  const int wave = tid >> 5;
  const int row0 = blockIdx.x * ROWS_BLK;               // chunk-row base
  const float4* chunk4 = (const float4*)chunk;

  // Per-lane query slice: columns lane*16 .. lane*16+15
  const float4* q4 = (const float4*)query + lane * 4;
  float4 q0 = q4[0], q1 = q4[1], q2 = q4[2], q3 = q4[3];

  // ---- Phase A: scores + stage chunk tile to LDS as f16 (single HBM read) -
  #pragma unroll 2
  for (int r = 0; r < 16; ++r) {
    int lrow = wave * 16 + r;
    const float4* p = chunk4 + (row0 + lrow) * (D_MODEL / 4) + lane * 4;
    float4 x0 = p[0], x1 = p[1], x2 = p[2], x3 = p[3];
    float s = q0.x*x0.x + q0.y*x0.y + q0.z*x0.z + q0.w*x0.w
            + q1.x*x1.x + q1.y*x1.y + q1.z*x1.z + q1.w*x1.w
            + q2.x*x2.x + q2.y*x2.y + q2.z*x2.z + q2.w*x2.w
            + q3.x*x3.x + q3.y*x3.y + q3.z*x3.z + q3.w*x3.w;
    v8h h0, h1;
    h0[0] = (_Float16)x0.x; h0[1] = (_Float16)x0.y;
    h0[2] = (_Float16)x0.z; h0[3] = (_Float16)x0.w;
    h0[4] = (_Float16)x1.x; h0[5] = (_Float16)x1.y;
    h0[6] = (_Float16)x1.z; h0[7] = (_Float16)x1.w;
    h1[0] = (_Float16)x2.x; h1[1] = (_Float16)x2.y;
    h1[2] = (_Float16)x2.z; h1[3] = (_Float16)x2.w;
    h1[4] = (_Float16)x3.x; h1[5] = (_Float16)x3.y;
    h1[6] = (_Float16)x3.z; h1[7] = (_Float16)x3.w;
    *(v8h*)&sC[lrow * D_MODEL + lane * 16]     = h0;
    *(v8h*)&sC[lrow * D_MODEL + lane * 16 + 8] = h1;
    #pragma unroll
    for (int off = 16; off > 0; off >>= 1) s += __shfl_xor(s, off, 32);
    if (lane == 0) sScores[lrow] = s * 0.0441941738241592f;   // 1/sqrt(512)
  }
  __syncthreads();

  // ---- Phase B: softmax over ratio=4 --------------------------------------
  if (tid < ROWS_BLK) {
    int g = tid >> 2, r = tid & 3;
    float s0 = sScores[g*4+0], s1 = sScores[g*4+1];
    float s2 = sScores[g*4+2], s3 = sScores[g*4+3];
    float m = fmaxf(fmaxf(s0, s1), fmaxf(s2, s3));
    float e0 = __expf(s0 - m), e1 = __expf(s1 - m);
    float e2 = __expf(s2 - m), e3 = __expf(s3 - m);
    float inv = 1.0f / (e0 + e1 + e2 + e3);
    float mine = (r == 0) ? e0 : (r == 1) ? e1 : (r == 2) ? e2 : e3;
    sAttn[tid] = mine * inv;
  }
  __syncthreads();

  // ---- Phase C: pool from LDS into registers, then write aliased A tile ---
  v4h hreg[16];
  #pragma unroll
  for (int i = 0; i < 16; ++i) {
    int idx = tid + 256 * i;          // 4096 v4h slots = 32 rows x 128
    int g  = idx >> 7;
    int c4 = idx & 127;
    float a0 = sAttn[g*4+0], a1 = sAttn[g*4+1];
    float a2 = sAttn[g*4+2], a3 = sAttn[g*4+3];
    int base = (g * 4) * D_MODEL + c4 * 4;
    v4h x0 = *(const v4h*)&sC[base];
    v4h x1 = *(const v4h*)&sC[base + D_MODEL];
    v4h x2 = *(const v4h*)&sC[base + 2 * D_MODEL];
    v4h x3 = *(const v4h*)&sC[base + 3 * D_MODEL];
    v4h h;
    h[0] = (_Float16)(a0*(float)x0[0] + a1*(float)x1[0] + a2*(float)x2[0] + a3*(float)x3[0]);
    h[1] = (_Float16)(a0*(float)x0[1] + a1*(float)x1[1] + a2*(float)x2[1] + a3*(float)x3[1]);
    h[2] = (_Float16)(a0*(float)x0[2] + a1*(float)x1[2] + a2*(float)x2[2] + a3*(float)x3[2]);
    h[3] = (_Float16)(a0*(float)x0[3] + a1*(float)x1[3] + a2*(float)x2[3] + a3*(float)x3[3]);
    hreg[i] = h;
  }
  __syncthreads();                     // staging reads done; safe to overwrite
  #pragma unroll
  for (int i = 0; i < 16; ++i) {
    int idx = tid + 256 * i;
    int g  = idx >> 7;
    int c4 = idx & 127;
    *(v4h*)&sA[g * A_LD + c4 * 4] = hreg[i];
  }
  __syncthreads();

  // ---- Phase D: out[32x512] = A[32x512] * w^T via v_wmma_f32_16x16x32_f16 -
  // 2 M-tiles x 2 N-tiles per wave iteration = 4 independent accumulator
  // chains (covers the ~5-slot WMMA->WMMA RAW latency).
  // Fragment octet pattern per ISA 7.12.2 (16-bit, K=32):
  //   lanes 0-15  hold K = {o..o+7, o+16..o+23};  lanes 16-31 the other octets.
  // B column n == w row n (contiguous), so A and B share the gather pattern.
  const int hh   = lane >> 4;
  const int colx = lane & 15;
  const _Float16* arow0 = &sA[colx * A_LD];
  const _Float16* arow1 = &sA[(16 + colx) * A_LD];
  #pragma unroll
  for (int j = 0; j < 2; ++j) {
    int nb0 = (wave + 16 * j) * 16;       // N-tile pair: wave+16j, wave+16j+8
    int nb1 = nb0 + 128;
    v8f acc00 = {}, acc01 = {}, acc10 = {}, acc11 = {};   // [m][n]
    const _Float16* wrow0 = wf16 + (nb0 + colx) * D_MODEL;
    const _Float16* wrow1 = wf16 + (nb1 + colx) * D_MODEL;
    #pragma unroll 2
    for (int kb = 0; kb < NKB; ++kb) {
      int o0 = kb * KBLK + hh * 8;
      int o1 = o0 + 16;
      v8h b0lo = *(const v8h*)(wrow0 + o0);
      v8h b0hi = *(const v8h*)(wrow0 + o1);
      v8h b1lo = *(const v8h*)(wrow1 + o0);
      v8h b1hi = *(const v8h*)(wrow1 + o1);
      v8h a0lo = *(const v8h*)(arow0 + o0);
      v8h a0hi = *(const v8h*)(arow0 + o1);
      v8h a1lo = *(const v8h*)(arow1 + o0);
      v8h a1hi = *(const v8h*)(arow1 + o1);
      v16h bf0, bf1, af0, af1;
      #pragma unroll
      for (int q = 0; q < 8; ++q) {
        bf0[q] = b0lo[q]; bf0[8+q] = b0hi[q];
        bf1[q] = b1lo[q]; bf1[8+q] = b1hi[q];
        af0[q] = a0lo[q]; af0[8+q] = a0hi[q];
        af1[q] = a1lo[q]; af1[8+q] = a1hi[q];
      }
      acc00 = __builtin_amdgcn_wmma_f32_16x16x32_f16(false, af0, false, bf0,
                                                     (short)0, acc00, false, false);
      acc10 = __builtin_amdgcn_wmma_f32_16x16x32_f16(false, af1, false, bf0,
                                                     (short)0, acc10, false, false);
      acc01 = __builtin_amdgcn_wmma_f32_16x16x32_f16(false, af0, false, bf1,
                                                     (short)0, acc01, false, false);
      acc11 = __builtin_amdgcn_wmma_f32_16x16x32_f16(false, af1, false, bf1,
                                                     (short)0, acc11, false, false);
    }
    // D layout: lane -> n = lane&15; m = (lane>>4)*8 + vgpr
    float bn0 = bias[nb0 + colx];
    float bn1 = bias[nb1 + colx];
    int gout = blockIdx.x * M_BLK;
    #pragma unroll
    for (int v = 0; v < 8; ++v) {
      int m = hh * 8 + v;
      __builtin_nontemporal_store(acc00[v] + bn0,
          &out[(gout + m) * D_MODEL + nb0 + colx]);
      __builtin_nontemporal_store(acc10[v] + bn0,
          &out[(gout + 16 + m) * D_MODEL + nb0 + colx]);
      __builtin_nontemporal_store(acc01[v] + bn1,
          &out[(gout + m) * D_MODEL + nb1 + colx]);
      __builtin_nontemporal_store(acc11[v] + bn1,
          &out[(gout + 16 + m) * D_MODEL + nb1 + colx]);
    }
  }
}

extern "C" void kernel_launch(void* const* d_in, const int* in_sizes, int n_in,
                              void* d_out, int out_size, void* d_ws, size_t ws_size,
                              hipStream_t stream) {
  const float* chunk = (const float*)d_in[0];   // [262144, 512] f32
  const float* query = (const float*)d_in[1];   // [512] f32
  const float* w     = (const float*)d_in[2];   // [512, 512] f32
  const float* b     = (const float*)d_in[3];   // [512] f32
  float* out = (float*)d_out;                   // [65536, 512] f32
  _Float16* wf16 = (_Float16*)d_ws;             // 512 KB workspace

  // 512*512 elems / (256 thr * 4 elems) = 256 blocks
  cvt_w_f16<<<256, 256, 0, stream>>>(w, wf16);

  const int n_groups = in_sizes[0] / D_MODEL / RATIO;   // 65536
  attnpool_wmma<<<n_groups / M_BLK, 256, 0, stream>>>(chunk, query, wf16, b, out);
}
